// SpeechResidualNet_53781580480541
// MI455X (gfx1250) — matrix operand order repaired and checked by reading, not compiled, not run on hardware
//
#include <hip/hip_runtime.h>
#include <hip/hip_bf16.h>

typedef _Float16 v16h __attribute__((ext_vector_type(16)));
typedef float    v8f  __attribute__((ext_vector_type(8)));

#define TPB       256
#define B_SZ      4
#define T_SZ      8192
#define RES_CH    64
#define DIL_CH    128
#define SKIP_CH   128
#define COND_DIM  80
#define TILE_T    64
#define K1        208          // 64 (tap0) + 64 (tap1) + 80 (cond)
#define K1P       224          // padded to 7 * 32
#define KSTEP1    7
#define XA_STRIDE 232          // halves per time-column (pad vs bank conflicts)
#define KSTEP2    2
#define ZB_STRIDE 72           // halves per time-column for z
#define H_STRIDE  72           // halves per row for activated h

// ---- weight accessors (per-layer base pointers) ----
__device__ __forceinline__ float w1_at(const float* __restrict__ convw,
                                       const float* __restrict__ condw,
                                       int o, int k) {
  if (k < 64)       return convw[o * (RES_CH * 2) + k * 2 + 0];        // tap f=0 -> x[t-d]
  else if (k < 128) return convw[o * (RES_CH * 2) + (k - 64) * 2 + 1]; // tap f=1 -> x[t]
  else if (k < K1)  return condw[o * COND_DIM + (k - 128)];
  return 0.0f;
}

__device__ __forceinline__ float w2_at(const float* __restrict__ resw,
                                       const float* __restrict__ skipw,
                                       int r, int k) {
  return (r < RES_CH) ? resw[r * 64 + k] : skipw[(r - RES_CH) * 64 + k];
}

// A-fragment (16x32 f16, M x K): lane m = row; kb = 8*(lane>=16);
// halves 0..15 -> K0+2*kb+0..15
__device__ __forceinline__ v16h load_afrag1(const float* __restrict__ convw,
                                            const float* __restrict__ condw,
                                            int m, int K0, int kb) {
  v16h a;
#pragma unroll
  for (int i = 0; i < 8; ++i) {
    a[i]     = (_Float16)w1_at(convw, condw, m, K0 + 2 * kb + i);
    a[8 + i] = (_Float16)w1_at(convw, condw, m, K0 + 2 * kb + 8 + i);
  }
  return a;
}

__device__ __forceinline__ v16h load_afrag2(const float* __restrict__ resw,
                                            const float* __restrict__ skipw,
                                            int m, int K0, int kb) {
  v16h a;
#pragma unroll
  for (int i = 0; i < 8; ++i) {
    a[i]     = (_Float16)w2_at(resw, skipw, m, K0 + 2 * kb + i);
    a[8 + i] = (_Float16)w2_at(resw, skipw, m, K0 + 2 * kb + 8 + i);
  }
  return a;
}

// B-fragment (32x16 f16, K x N) from column-major LDS tile:
// lane n = col (lane&15); koff = K0 + 16*(lane>=16); 16 contiguous halves.
__device__ __forceinline__ v16h loadB16(const _Float16* p0, int stride, int col, int koff) {
  union { v16h v; uint4 u[2]; } t;
  const _Float16* p = p0 + col * stride + koff;
  t.u[0] = *(const uint4*)(p);
  t.u[1] = *(const uint4*)(p + 8);
  return t.v;
}

__device__ __forceinline__ float fast_sigmoid(float x) { return 1.0f / (1.0f + __expf(-x)); }
__device__ __forceinline__ float fast_tanh(float x)    { return 2.0f / (1.0f + __expf(-2.0f * x)) - 1.0f; }

__global__ __launch_bounds__(TPB) void wavenet_layer_kernel(
    const float* __restrict__ xs_in, float* __restrict__ xs_out,
    const float* __restrict__ cond,  float* __restrict__ skip_out,
    const float* __restrict__ convw, const float* __restrict__ convb,
    const float* __restrict__ condw, const float* __restrict__ condb,
    const float* __restrict__ resw,  const float* __restrict__ resb,
    const float* __restrict__ skipw, const float* __restrict__ skipb,
    int dil, int first) {
  __shared__ __align__(16) _Float16 sXa[TILE_T * XA_STRIDE]; // 29,696 B (col-major, K padded+zeroed)
  __shared__ __align__(16) _Float16 sH [DIL_CH * H_STRIDE];  // 18,432 B (activated h, row-major)
  __shared__ __align__(16) _Float16 sZ [TILE_T * ZB_STRIDE]; //  9,216 B (z, col-major)

  const int tid    = threadIdx.x;
  const int wid    = tid >> 5;
  const int lane   = tid & 31;
  const int lanelo = lane & 15;
  const int lanehi = lane >> 4;
  const int blk    = blockIdx.x;
  const int b      = blk >> 7;            // 128 tiles per batch
  const int t0     = (blk & 127) * TILE_T;

  // ---------------- stage Xa (f32 -> f16), 4 consecutive K per thread ----------------
#pragma unroll
  for (int it = 0; it < (K1P / 4) * TILE_T / TPB; ++it) {   // 14 iterations
    int idx = it * TPB + tid;
    int j   = idx & (TILE_T - 1);
    int k4  = (idx >> 6) * 4;
    float f[4];
    if (k4 < RES_CH) {                 // x[t - d]
      int t = t0 + j - dil;
#pragma unroll
      for (int r = 0; r < 4; ++r)
        f[r] = (t >= 0) ? xs_in[((size_t)b * RES_CH + (k4 + r)) * T_SZ + t] : 0.0f;
    } else if (k4 < 2 * RES_CH) {      // x[t]
      int t = t0 + j;
#pragma unroll
      for (int r = 0; r < 4; ++r)
        f[r] = xs_in[((size_t)b * RES_CH + (k4 - 64 + r)) * T_SZ + t];
    } else if (k4 < K1) {              // cond[t]
      int t = t0 + j;
#pragma unroll
      for (int r = 0; r < 4; ++r)
        f[r] = cond[((size_t)b * COND_DIM + (k4 - 128 + r)) * T_SZ + t];
    } else {
#pragma unroll
      for (int r = 0; r < 4; ++r) f[r] = 0.0f;
    }
    union { _Float16 h[4]; unsigned long long u64; } p;
#pragma unroll
    for (int r = 0; r < 4; ++r) p.h[r] = (_Float16)f[r];
    *(unsigned long long*)&sXa[j * XA_STRIDE + k4] = p.u64;
  }

  // ---------------- A fragments + biases in registers ----------------
  v16h a1[KSTEP1];
  {
    int m = wid * 16 + lanelo;
#pragma unroll
    for (int ks = 0; ks < KSTEP1; ++ks)
      a1[ks] = load_afrag1(convw, condw, m, ks * 32, lanehi * 8);
  }
  float cbias[8];
#pragma unroll
  for (int v = 0; v < 8; ++v) {
    int row  = wid * 16 + lanehi * 8 + v;
    cbias[v] = convb[row] + condb[row];
  }

  const int nT2 = (wid < 4) ? 2 : 1;   // waves 0-3 own res tile w and skip tile 8+w
  v16h  a2[2][KSTEP2];
  float bias2[2][8];
#pragma unroll
  for (int tix = 0; tix < 2; ++tix) {
    if (tix >= nT2) continue;
    int mt = tix ? (8 + wid) : wid;
    int m  = mt * 16 + lanelo;
#pragma unroll
    for (int ks = 0; ks < KSTEP2; ++ks)
      a2[tix][ks] = load_afrag2(resw, skipw, m, ks * 32, lanehi * 8);
#pragma unroll
    for (int v = 0; v < 8; ++v) {
      int row = mt * 16 + lanehi * 8 + v;
      bias2[tix][v] = (mt < 4) ? resb[row] : skipb[row - 64];
    }
  }

  __syncthreads();

  // ---------------- GEMM1: h = W_a * Xa  (wave w -> M-tile w) ----------------
#pragma unroll
  for (int nt = 0; nt < 4; ++nt) {
    v8f c;
#pragma unroll
    for (int v = 0; v < 8; ++v) c[v] = cbias[v];
#pragma unroll
    for (int ks = 0; ks < KSTEP1; ++ks) {
      v16h bf = loadB16(sXa, XA_STRIDE, nt * 16 + lanelo, ks * 32 + lanehi * 16);
      c = __builtin_amdgcn_wmma_f32_16x16x32_f16(false, a1[ks], false, bf,
                                                 (short)0, c, false, false);
    }
    // gated-activation halves: waves 0-3 hold tanh rows, waves 4-7 sigmoid rows
#pragma unroll
    for (int v = 0; v < 8; ++v) {
      int row   = wid * 16 + lanehi * 8 + v;
      float hv  = c[v];
      float act = (wid < 4) ? fast_tanh(hv) : fast_sigmoid(hv);
      sH[row * H_STRIDE + nt * 16 + lanelo] = (_Float16)act;
    }
  }
  __syncthreads();

  // ---------------- z = tanh * sigmoid, staged col-major for GEMM2 ----------------
#pragma unroll
  for (int it = 0; it < (RES_CH / 4) * TILE_T / TPB; ++it) {  // 4 iterations
    int idx = it * TPB + tid;
    int j   = idx & (TILE_T - 1);
    int c4  = (idx >> 6) * 4;
    union { _Float16 h[4]; unsigned long long u64; } p;
#pragma unroll
    for (int r = 0; r < 4; ++r) {
      int ch = c4 + r;
      float z = (float)sH[ch * H_STRIDE + j] * (float)sH[(ch + 64) * H_STRIDE + j];
      p.h[r] = (_Float16)z;
    }
    *(unsigned long long*)&sZ[j * ZB_STRIDE + c4] = p.u64;
  }
  __syncthreads();

  // ---------------- GEMM2: [res;skip] = W_b * z, + residual / skip accumulate ----------------
#pragma unroll
  for (int nt = 0; nt < 4; ++nt) {
    v16h b0 = loadB16(sZ, ZB_STRIDE, nt * 16 + lanelo, 0  + lanehi * 16);
    v16h b1 = loadB16(sZ, ZB_STRIDE, nt * 16 + lanelo, 32 + lanehi * 16);
#pragma unroll
    for (int tix = 0; tix < 2; ++tix) {
      if (tix >= nT2) continue;
      v8f c;
#pragma unroll
      for (int v = 0; v < 8; ++v) c[v] = 0.0f;
      c = __builtin_amdgcn_wmma_f32_16x16x32_f16(false, a2[tix][0], false, b0,
                                                 (short)0, c, false, false);
      c = __builtin_amdgcn_wmma_f32_16x16x32_f16(false, a2[tix][1], false, b1,
                                                 (short)0, c, false, false);
      int mt = tix ? (8 + wid) : wid;
      int t  = t0 + nt * 16 + lanelo;
      if (mt < 4) {                      // residual rows 0..63
#pragma unroll
        for (int v = 0; v < 8; ++v) {
          int o = mt * 16 + lanehi * 8 + v;
          size_t gi = ((size_t)b * RES_CH + o) * T_SZ + t;
          xs_out[gi] = c[v] + bias2[tix][v] + xs_in[gi];
        }
      } else {                           // skip rows 64..191 -> skip ch 0..127
#pragma unroll
        for (int v = 0; v < 8; ++v) {
          int o = mt * 16 + lanehi * 8 + v - 64;
          size_t gi = ((size_t)b * SKIP_CH + o) * T_SZ + t;
          float prev = first ? 0.0f : skip_out[gi];
          skip_out[gi] = prev + c[v] + bias2[tix][v];
        }
      }
    }
  }
}

extern "C" void kernel_launch(void* const* d_in, const int* in_sizes, int n_in,
                              void* d_out, int out_size, void* d_ws, size_t ws_size,
                              hipStream_t stream) {
  (void)in_sizes; (void)n_in; (void)out_size; (void)ws_size;
  const float* x     = (const float*)d_in[0];
  const float* cond  = (const float*)d_in[1];
  const float* convw = (const float*)d_in[2];
  const float* convb = (const float*)d_in[3];
  const float* condw = (const float*)d_in[4];
  const float* condb = (const float*)d_in[5];
  const float* resw  = (const float*)d_in[6];
  const float* resb  = (const float*)d_in[7];
  const float* skipw = (const float*)d_in[8];
  const float* skipb = (const float*)d_in[9];
  float* skip_out = (float*)d_out;

  const size_t xsElems = (size_t)B_SZ * RES_CH * T_SZ;
  float* bufA = (float*)d_ws;            // xs ping-pong buffers
  float* bufB = bufA + xsElems;

  dim3 grid(B_SZ * (T_SZ / TILE_T));     // 512 blocks
  for (int i = 0; i < 20; ++i) {
    int d = 1 << (i % 10);
    const float* xin = (i == 0) ? x : ((i & 1) ? bufA : bufB);
    float*       xout = (i & 1) ? bufB : bufA;
    wavenet_layer_kernel<<<grid, TPB, 0, stream>>>(
        xin, xout, cond, skip_out,
        convw + (size_t)i * DIL_CH * RES_CH * 2, convb + (size_t)i * DIL_CH,
        condw + (size_t)i * DIL_CH * COND_DIM,   condb + (size_t)i * DIL_CH,
        resw  + (size_t)i * RES_CH * (DIL_CH/2), resb  + (size_t)i * RES_CH,
        skipw + (size_t)i * SKIP_CH * (DIL_CH/2), skipb + (size_t)i * SKIP_CH,
        d, (i == 0) ? 1 : 0);
  }
}
